// PointnetFPModule_16776142258206
// MI455X (gfx1250) — matrix operand order repaired and checked by reading, not compiled, not run on hardware
//
#include <hip/hip_runtime.h>
#include <hip/hip_bf16.h>

// ---------------------------------------------------------------------------
// PointNet++ FP module for MI455X (gfx1250, wave32, WMMA + TDM)
//   B=16, N=4096, M=1024, C1=256, C2=512, MLP 768->256->256
// GEMMs: v_wmma_f32_16x16x32_f16, weight tiles staged by tensor_load_to_lds.
// ---------------------------------------------------------------------------

typedef _Float16 f16;
typedef __attribute__((ext_vector_type(8)))  _Float16 v8h;
typedef __attribute__((ext_vector_type(16))) _Float16 v16h;
typedef __attribute__((ext_vector_type(8)))  float    v8f;
typedef __attribute__((ext_vector_type(4)))  unsigned int u32x4;
typedef __attribute__((ext_vector_type(8)))  int      i32x8;
typedef __attribute__((ext_vector_type(4)))  int      i32x4;

#define BATCH   16
#define NQ      4096
#define MK      1024
#define C_IN1   768
#define C_MID   256
#define ROWS    (BATCH * NQ)          // 65536
#define BN_EPS  1e-5f

#if defined(__gfx1250__) && __has_builtin(__builtin_amdgcn_tensor_load_to_lds)
#define USE_TDM 1
#else
#define USE_TDM 0
#endif

// ---------------------------------------------------------------------------
// 1) Brute-force 3-NN: known xyz staged in LDS, per-thread top-3 insertion.
//    grid = (NQ/256, BATCH), block = 256
// ---------------------------------------------------------------------------
__global__ void knn_kernel(const float* __restrict__ unknown,  // (B,N,3)
                           const float* __restrict__ known,    // (B,M,3)
                           int*  __restrict__ idx_out,         // (B*N,3)
                           float* __restrict__ w_out) {        // (B*N,3)
  __shared__ float kx[MK], ky[MK], kz[MK];
  const int b = blockIdx.y;
  const int n = blockIdx.x * 256 + threadIdx.x;
  const float* kb = known + (size_t)b * MK * 3;
  for (int i = threadIdx.x; i < MK; i += 256) {
    kx[i] = kb[i * 3 + 0];
    ky[i] = kb[i * 3 + 1];
    kz[i] = kb[i * 3 + 2];
  }
  __syncthreads();

  const float* u = unknown + ((size_t)b * NQ + n) * 3;
  const float ux = u[0], uy = u[1], uz = u[2];

  float d0 = 1e30f, d1 = 1e30f, d2 = 1e30f;
  int   i0 = 0,     i1 = 0,     i2 = 0;
  #pragma unroll 4
  for (int m = 0; m < MK; ++m) {
    const float dx = ux - kx[m], dy = uy - ky[m], dz = uz - kz[m];
    const float d = dx * dx + dy * dy + dz * dz;
    if (d < d0)      { d2 = d1; i2 = i1; d1 = d0; i1 = i0; d0 = d; i0 = m; }
    else if (d < d1) { d2 = d1; i2 = i1; d1 = d;  i1 = m; }
    else if (d < d2) { d2 = d;  i2 = m; }
  }
  const float w0 = 1.0f / (d0 + 1e-10f);
  const float w1 = 1.0f / (d1 + 1e-10f);
  const float w2 = 1.0f / (d2 + 1e-10f);
  const float inv = 1.0f / (w0 + w1 + w2);

  const size_t q = (size_t)b * NQ + n;
  idx_out[q * 3 + 0] = i0;
  idx_out[q * 3 + 1] = i1;
  idx_out[q * 3 + 2] = i2;
  w_out[q * 3 + 0] = w0 * inv;
  w_out[q * 3 + 1] = w1 * inv;
  w_out[q * 3 + 2] = w2 * inv;
}

// ---------------------------------------------------------------------------
// 2) Interpolate (B,n,512) + concat (B,n,256) -> X0 f16 (ROWS x 768)
//    grid = ROWS, block = 128
// ---------------------------------------------------------------------------
__global__ void interp_concat_kernel(const float* __restrict__ known_feats,  // (B,M,512)
                                     const float* __restrict__ unknow_feats, // (B,N,256)
                                     const int*   __restrict__ idx,
                                     const float* __restrict__ w,
                                     f16* __restrict__ X0) {
  const size_t q = blockIdx.x;
  const int b = (int)(q >> 12);            // q / 4096
  const int t = threadIdx.x;

  const int   i0 = idx[q * 3 + 0], i1 = idx[q * 3 + 1], i2 = idx[q * 3 + 2];
  const float w0 = w[q * 3 + 0],   w1 = w[q * 3 + 1],   w2 = w[q * 3 + 2];

  const float* f0 = known_feats + ((size_t)b * MK + i0) * 512;
  const float* f1 = known_feats + ((size_t)b * MK + i1) * 512;
  const float* f2 = known_feats + ((size_t)b * MK + i2) * 512;
  f16* xr = X0 + q * C_IN1;

  #pragma unroll
  for (int c = t; c < 512; c += 128)
    xr[c] = (f16)(w0 * f0[c] + w1 * f1[c] + w2 * f2[c]);

  const float* uf = unknow_feats + q * 256;
  #pragma unroll
  for (int c = t; c < 256; c += 128)
    xr[512 + c] = (f16)uf[c];
}

// ---------------------------------------------------------------------------
// 3) Weight fp32 -> f16 (keeps (Cout,Cin) = N-major x K layout)
// ---------------------------------------------------------------------------
__global__ void wcvt_kernel(const float* __restrict__ W, f16* __restrict__ Wh, int count) {
  const int e = blockIdx.x * blockDim.x + threadIdx.x;
  if (e < count) Wh[e] = (f16)W[e];
}

// ---------------------------------------------------------------------------
// 4) WMMA GEMM: C[M x N] = A[M x K] * Bw[N x K]^T   (f16 inputs, f32 accum)
//    Block tile 128(M) x 128(N), 256 threads = 8 waves (4 m-waves x 2 n-waves),
//    each wave: 32x64 strip = 2x4 v_wmma_f32_16x16x32_f16 per K-step of 32.
//    A tile staged manually (b128 loads -> ds b128 stores); B (weight) tile
//    staged by the Tensor Data Mover when available (tensor_load_to_lds with a
//    2D D#: tile 32(K) x 128(N), data_size=2B, dim0_stride=K), completion via
//    s_wait_tensorcnt before the workgroup barrier.
//    grid = (M/128, N/128)
// ---------------------------------------------------------------------------
__global__ void gemm_f16_wmma(const f16* __restrict__ A,
                              const f16* __restrict__ Bw,
                              float* __restrict__ C,
                              int K, int N) {
  __shared__ __align__(16) f16 smem[128 * 32 + 128 * 32];
  f16* sA = smem;              // 128 rows x 32 halves  (8 KB)
  f16* sB = smem + 128 * 32;   // 128 rows x 32 halves  (8 KB), N-major

  const int tid  = threadIdx.x;
  const int lane = tid & 31;
  const int wid  = tid >> 5;
  const int wm   = wid & 3;              // 0..3 : which 32-row strip
  const int wn   = wid >> 2;             // 0..1 : which 64-col strip
  const int lrow = lane & 15;
  const int lhi  = lane >> 4;

  const size_t m0 = (size_t)blockIdx.x * 128;
  const int    n0 = blockIdx.y * 128;

  // staging: 128x32 halves = 512 chunks of 8 halves (16B); 2 chunks/thread
  const int ca0 = tid * 2;
  const int ar0 = ca0 >> 2,       ac0 = (ca0 & 3) * 8;
  const int ar1 = (ca0 + 1) >> 2, ac1 = ((ca0 + 1) & 3) * 8;

  v8f acc[2][4];
  #pragma unroll
  for (int i = 0; i < 2; ++i)
    #pragma unroll
    for (int j = 0; j < 4; ++j) acc[i][j] = v8f{};

  for (int kt = 0; kt < K; kt += 32) {
    if (kt + 32 < K) {  // uniform branch: EXEC stays all-ones for WMMA
      __builtin_prefetch(A + (m0 + ar0) * (size_t)K + kt + 32 + ac0, 0, 3);
      __builtin_prefetch(A + (m0 + ar1) * (size_t)K + kt + 32 + ac1, 0, 3);
    }
    __syncthreads();

    // ---- stage A tile: 128 x 32 halves ----
    *(v8h*)(sA + ar0 * 32 + ac0) =
        *(const v8h*)(A + (m0 + ar0) * (size_t)K + kt + ac0);
    *(v8h*)(sA + ar1 * 32 + ac1) =
        *(const v8h*)(A + (m0 + ar1) * (size_t)K + kt + ac1);

    // ---- stage B tile: 128 x 32 halves (N-major, K contiguous) ----
#if USE_TDM
    if (wid == 0) {  // wave-uniform; TDM ignores EXEC, one issue per block
      const unsigned long long gaddr =
          (unsigned long long)(const void*)(Bw + (size_t)n0 * K + kt);
      u32x4 g0;
      g0.x = 1u;                                   // count=1, user mode
      g0.y = (unsigned)(128 * 32 * 2);             // lds byte addr of sB
      g0.z = (unsigned)(gaddr & 0xFFFFFFFFu);      // global_addr[31:0]
      g0.w = (unsigned)((gaddr >> 32) & 0x1FFFFFFu) | (2u << 30);  // type=2
      i32x8 g1;
      g1[0] = (1 << 16);                           // data_size=1 (2 bytes)
      g1[1] = (int)((K & 0xFFFF) << 16);           // tensor_dim0[15:0]
      g1[2] = (int)(((unsigned)K >> 16) & 0xFFFFu) | ((N & 0xFFFF) << 16);
      g1[3] = (int)(((unsigned)N >> 16) & 0xFFFFu) | (32 << 16);   // tile_dim0=32
      g1[4] = 128;                                 // tile_dim1=128, tile_dim2=0
      g1[5] = K;                                   // tensor_dim0_stride[31:0]
      g1[6] = 0;                                   // stride hi / dim1_stride lo
      g1[7] = 0;
      const i32x4 z4 = {0, 0, 0, 0};
#if __has_include(<hip/amd_detail/amd_gfx1250_TDM.h>)
      const i32x8 z8 = {0, 0, 0, 0, 0, 0, 0, 0};
      __builtin_amdgcn_tensor_load_to_lds(g0, g1, z4, z4, z8, 0);
#else
      __builtin_amdgcn_tensor_load_to_lds(g0, g1, z4, z4, 0);
#endif
      __builtin_amdgcn_s_wait_tensorcnt(0);
    }
#else
    {
      const int cb0 = tid * 2;
      const int br0 = cb0 >> 2,       bc0 = (cb0 & 3) * 8;
      const int br1 = (cb0 + 1) >> 2, bc1 = ((cb0 + 1) & 3) * 8;
      *(v8h*)(sB + br0 * 32 + bc0) =
          *(const v8h*)(Bw + (size_t)(n0 + br0) * K + kt + bc0);
      *(v8h*)(sB + br1 * 32 + bc1) =
          *(const v8h*)(Bw + (size_t)(n0 + br1) * K + kt + bc1);
    }
#endif
    __syncthreads();

    // ---- A fragments (16x32 f16), per ISA lane layout ----
    const f16* ap0 = sA + (wm * 32 + lrow) * 32 + lhi * 8;
    const f16* ap1 = sA + (wm * 32 + 16 + lrow) * 32 + lhi * 8;
    const v8h a0lo = *(const v8h*)ap0;
    const v8h a0hi = *(const v8h*)(ap0 + 16);
    const v8h a1lo = *(const v8h*)ap1;
    const v8h a1hi = *(const v8h*)(ap1 + 16);
    const v16h af0 = __builtin_shufflevector(
        a0lo, a0hi, 0, 1, 2, 3, 4, 5, 6, 7, 8, 9, 10, 11, 12, 13, 14, 15);
    const v16h af1 = __builtin_shufflevector(
        a1lo, a1hi, 0, 1, 2, 3, 4, 5, 6, 7, 8, 9, 10, 11, 12, 13, 14, 15);

    #pragma unroll
    for (int j = 0; j < 4; ++j) {
      // B fragment (32x16 f16): lane n = lrow, K = lhi*8..+7 then 16+lhi*8..+7
      const f16* bp = sB + (wn * 64 + j * 16 + lrow) * 32 + lhi * 8;
      const v8h blo = *(const v8h*)bp;
      const v8h bhi = *(const v8h*)(bp + 16);
      const v16h bf = __builtin_shufflevector(
          blo, bhi, 0, 1, 2, 3, 4, 5, 6, 7, 8, 9, 10, 11, 12, 13, 14, 15);
      acc[0][j] = __builtin_amdgcn_wmma_f32_16x16x32_f16(
          false, af0, false, bf, (short)0, acc[0][j], false, false);
      acc[1][j] = __builtin_amdgcn_wmma_f32_16x16x32_f16(
          false, af1, false, bf, (short)0, acc[1][j], false, false);
    }
  }

  // C/D layout: VGPR i, lanes 0-15 -> M=i, lanes 16-31 -> M=8+i; N = lane&15
  #pragma unroll
  for (int fi = 0; fi < 2; ++fi) {
    const size_t mrow = m0 + wm * 32 + fi * 16 + lhi * 8;
    #pragma unroll
    for (int j = 0; j < 4; ++j) {
      const int nc = n0 + wn * 64 + j * 16 + lrow;
      #pragma unroll
      for (int i = 0; i < 8; ++i)
        C[(mrow + i) * (size_t)N + nc] = acc[fi][j][i];
    }
  }
}

// ---------------------------------------------------------------------------
// 5) Column sum / sum-of-squares (training-mode BN stats), 256 channels.
//    grid = 256 blocks x 256 threads; thread t owns channel t over a row chunk.
// ---------------------------------------------------------------------------
__global__ void colstats_kernel(const float* __restrict__ Y,
                                float* __restrict__ sums,
                                float* __restrict__ sqs,
                                int rows) {
  const int c = threadIdx.x;                 // channel (coalesced across lane)
  const int chunk = rows / gridDim.x;
  const int r0 = blockIdx.x * chunk;
  float s = 0.f, s2 = 0.f;
  for (int r = r0; r < r0 + chunk; ++r) {
    const float v = Y[(size_t)r * 256 + c];
    s += v;
    s2 += v * v;
  }
  atomicAdd(&sums[c], s);
  atomicAdd(&sqs[c], s2);
}

// ---------------------------------------------------------------------------
// 6) BN (training stats) + ReLU -> f16 (feeds GEMM2) or f32 (final output)
// ---------------------------------------------------------------------------
__global__ void bn_relu_f16_kernel(const float* __restrict__ Y,
                                   const float* __restrict__ sums,
                                   const float* __restrict__ sqs,
                                   const float* __restrict__ gamma,
                                   const float* __restrict__ beta,
                                   f16* __restrict__ X, int total, float invRows) {
  const int e = blockIdx.x * blockDim.x + threadIdx.x;
  if (e >= total) return;
  const int c = e & 255;
  const float mean = sums[c] * invRows;
  const float var  = sqs[c] * invRows - mean * mean;
  const float sc   = rsqrtf(var + BN_EPS) * gamma[c];
  const float v    = (Y[e] - mean) * sc + beta[c];
  X[e] = (f16)fmaxf(v, 0.f);
}

__global__ void bn_relu_f32_kernel(const float* __restrict__ Y,
                                   const float* __restrict__ sums,
                                   const float* __restrict__ sqs,
                                   const float* __restrict__ gamma,
                                   const float* __restrict__ beta,
                                   float* __restrict__ O, int total, float invRows) {
  const int e = blockIdx.x * blockDim.x + threadIdx.x;
  if (e >= total) return;
  const int c = e & 255;
  const float mean = sums[c] * invRows;
  const float var  = sqs[c] * invRows - mean * mean;
  const float sc   = rsqrtf(var + BN_EPS) * gamma[c];
  const float v    = (Y[e] - mean) * sc + beta[c];
  O[e] = fmaxf(v, 0.f);
}

// ---------------------------------------------------------------------------
// Launch
// ---------------------------------------------------------------------------
extern "C" void kernel_launch(void* const* d_in, const int* in_sizes, int n_in,
                              void* d_out, int out_size, void* d_ws, size_t ws_size,
                              hipStream_t stream) {
  (void)in_sizes; (void)n_in; (void)out_size; (void)ws_size;

  const float* unknown      = (const float*)d_in[0];  // (B,N,3)
  const float* known        = (const float*)d_in[1];  // (B,M,3)
  const float* unknow_feats = (const float*)d_in[2];  // (B,N,256)
  const float* known_feats  = (const float*)d_in[3];  // (B,M,512)
  const float* W1     = (const float*)d_in[4];        // (256,768)
  const float* gamma1 = (const float*)d_in[5];
  const float* beta1  = (const float*)d_in[6];
  const float* W2     = (const float*)d_in[7];        // (256,256)
  const float* gamma2 = (const float*)d_in[8];
  const float* beta2  = (const float*)d_in[9];
  float* out = (float*)d_out;

  char* ws = (char*)d_ws;
  // Workspace layout (regions reused once their producer/consumer are done):
  const size_t OFF_X0  = 0;                               // f16 ROWSx768 (96 MB); reused for X2
  const size_t OFF_Y1  = OFF_X0 + (size_t)ROWS * C_IN1 * 2;   // f32 ROWSx256 (64 MB); reused for Y2
  const size_t OFF_W1H = OFF_Y1 + (size_t)ROWS * C_MID * 4;
  const size_t OFF_W2H = OFF_W1H + (size_t)C_MID * C_IN1 * 2;
  const size_t OFF_IDX = OFF_W2H + (size_t)C_MID * C_MID * 2;
  const size_t OFF_WGT = OFF_IDX + (size_t)ROWS * 3 * 4;
  const size_t OFF_ST  = OFF_WGT + (size_t)ROWS * 3 * 4;  // 4 x 256 floats

  f16*   X0   = (f16*)(ws + OFF_X0);
  float* Y1   = (float*)(ws + OFF_Y1);
  f16*   X2   = (f16*)(ws + OFF_X0);   // reuse X0 region
  float* Y2   = (float*)(ws + OFF_Y1); // reuse Y1 region
  f16*   W1h  = (f16*)(ws + OFF_W1H);
  f16*   W2h  = (f16*)(ws + OFF_W2H);
  int*   idx  = (int*)(ws + OFF_IDX);
  float* wgt  = (float*)(ws + OFF_WGT);
  float* sum1 = (float*)(ws + OFF_ST);
  float* sq1  = sum1 + 256;
  float* sum2 = sum1 + 512;
  float* sq2  = sum1 + 768;

  const float invRows = 1.0f / (float)ROWS;

  hipMemsetAsync(sum1, 0, 4 * 256 * sizeof(float), stream);

  // 1) 3-NN indices + inverse-distance weights
  knn_kernel<<<dim3(NQ / 256, BATCH), 256, 0, stream>>>(unknown, known, idx, wgt);

  // weights to f16 (layout already N-major x K-contiguous)
  wcvt_kernel<<<(C_MID * C_IN1 + 255) / 256, 256, 0, stream>>>(W1, W1h, C_MID * C_IN1);
  wcvt_kernel<<<(C_MID * C_MID + 255) / 256, 256, 0, stream>>>(W2, W2h, C_MID * C_MID);

  // 2) interpolation + concat -> X0 f16 (ROWS x 768)
  interp_concat_kernel<<<ROWS, 128, 0, stream>>>(known_feats, unknow_feats, idx, wgt, X0);

  // 3) GEMM1 (WMMA): Y1 = X0 * W1^T   (65536x768 x 768->256)
  gemm_f16_wmma<<<dim3(ROWS / 128, C_MID / 128), 256, 0, stream>>>(X0, W1h, Y1, C_IN1, C_MID);

  // 4) BN1 stats + apply + ReLU -> X2 f16
  colstats_kernel<<<256, 256, 0, stream>>>(Y1, sum1, sq1, ROWS);
  bn_relu_f16_kernel<<<(ROWS * C_MID) / 256, 256, 0, stream>>>(
      Y1, sum1, sq1, gamma1, beta1, X2, ROWS * C_MID, invRows);

  // 5) GEMM2 (WMMA): Y2 = X2 * W2^T   (65536x256 x 256->256)
  gemm_f16_wmma<<<dim3(ROWS / 128, C_MID / 128), 256, 0, stream>>>(X2, W2h, Y2, C_MID, C_MID);

  // 6) BN2 stats + apply + ReLU -> fp32 output
  colstats_kernel<<<256, 256, 0, stream>>>(Y2, sum2, sq2, ROWS);
  bn_relu_f32_kernel<<<(ROWS * C_MID) / 256, 256, 0, stream>>>(
      Y2, sum2, sq2, gamma2, beta2, out, ROWS * C_MID, invRows);
}
